// FilterBasedTokenizer_37452114821261
// MI455X (gfx1250) — matrix operand order repaired and checked by reading, not compiled, not run on hardware
//
#include <hip/hip_runtime.h>

// ---------------------------------------------------------------------------
// Fused FilterBasedTokenizer for MI455X (gfx1250, wave32, WMMA bf16).
//
//   T[b,l,c] = ( sum_n exp(X[b,n,:]·wA[b,l,:]) * (X[b,n,:]·wV[b,:,c]) )
//            / ( sum_n exp(X[b,n,:]·wA[b,l,:]) )
//
// Single streaming pass over X (512 MB -> ~22 us @ 23.3 TB/s); all GEMMs on
// v_wmma_f32_16x16x32_bf16 with f32 accumulation. No softmax max-shift needed
// (logit std ~1.4, exp stays well inside f32 range; softmax is shift-invariant).
// Grid = 32 batches x 8 N-splits; partials reduced by a second tiny kernel.
// ---------------------------------------------------------------------------

#define Bd 32
#define Nd 16384
#define Dd 256
#define Ld 16
#define Cd 128
#define SPLITS 8
#define NSPL (Nd / SPLITS)   // 2048 rows of N per workgroup
#define BLKN 128             // n-rows staged per iteration
#define NITER (NSPL / BLKN)  // 16

typedef __attribute__((ext_vector_type(16))) __bf16 v16bf;
typedef __attribute__((ext_vector_type(8)))  float  v8f;
typedef __attribute__((ext_vector_type(4)))  unsigned u32x4;
typedef __attribute__((ext_vector_type(2)))  unsigned u32x2;

union Frag {
  v16bf v;
  u32x4 q[2];
  unsigned short s[16];
};

__device__ __forceinline__ unsigned short f2bf(float f) {
  union { float f; unsigned u; } x; x.f = f;
  unsigned r = x.u + 0x7FFFu + ((x.u >> 16) & 1u);   // round-to-nearest-even
  return (unsigned short)(r >> 16);
}

__global__ __launch_bounds__(256)
void tok_main(const float* __restrict__ X, const float* __restrict__ wA,
              const float* __restrict__ wV, float* __restrict__ Tp,
              float* __restrict__ Zp) {
  extern __shared__ char smem[];
  unsigned short* xs  = (unsigned short*)smem;   // [128][256] X tile, bf16
  unsigned short* wv  = xs  + 128 * 256;         // [256][128] wV,    bf16
  unsigned short* wa  = wv  + 256 * 128;         // [16][256]  wA,    bf16
  unsigned short* et  = wa  + 16 * 256;          // [16][128]  exp(A),bf16
  unsigned short* vvt = et  + 16 * 128;          // 8 x [16c][32n] VV^T scratch
  float*          zl  = (float*)(vvt + 8 * 16 * 32);  // [16] Z partial

  const int tid  = threadIdx.x;
  const int wave = tid >> 5;
  const int lane = tid & 31;
  const int hf   = lane >> 4;     // lane half (0/1)
  const int ln   = lane & 15;

  const int wg = blockIdx.x;
  const int b  = wg / SPLITS;
  const int sp = wg % SPLITS;

  const float* Xb  = X  + ((size_t)b * Nd + (size_t)sp * NSPL) * Dd;
  const float* wAb = wA + (size_t)b * Ld * Dd;
  const float* wVb = wV + (size_t)b * Dd * Cd;

  // ---- setup: stage wA and wV into LDS as bf16 (coalesced) ----
  for (int i = 0; i < (Ld * Dd) / 256; ++i) {
    int e = i * 256 + tid;
    wa[e] = f2bf(wAb[e]);
  }
  for (int i = 0; i < (Dd * Cd) / 256; ++i) {
    int e = i * 256 + tid;
    wv[e] = f2bf(wVb[e]);
  }
  if (tid < Ld) zl[tid] = 0.f;
  __syncthreads();

  // ---- persistent wV B-fragments for this wave's 16-wide c-tile ----
  // B layout (32x16 K x N): lane(hf, col=ln), v0..7 = K = 16*hf + 0..15
  Frag wvf[8];
  {
    const int c = wave * 16 + ln;
    for (int kc = 0; kc < 8; ++kc) {
#pragma unroll
      for (int j = 0; j < 8; ++j) {
        int k0 = kc * 32 + hf * 16 + 2 * j;
        wvf[kc].s[2 * j]     = wv[k0 * Cd + c];
        wvf[kc].s[2 * j + 1] = wv[(k0 + 1) * Cd + c];
      }
    }
  }

  v8f   tacc = {0.f, 0.f, 0.f, 0.f, 0.f, 0.f, 0.f, 0.f};
  float zacc[8] = {0.f, 0.f, 0.f, 0.f, 0.f, 0.f, 0.f, 0.f};

  for (int blk = 0; blk < NITER; ++blk) {
    // ---- stage 128x256 X tile into LDS as bf16 (float4 coalesced) ----
    const float4* X4 = (const float4*)(Xb + (size_t)blk * BLKN * Dd);
#pragma unroll 4
    for (int i = 0; i < 32; ++i) {
      float4 f = X4[i * 256 + tid];
      u32x2 p;
      p.x = (unsigned)f2bf(f.x) | ((unsigned)f2bf(f.y) << 16);
      p.y = (unsigned)f2bf(f.z) | ((unsigned)f2bf(f.w) << 16);
      *(u32x2*)(xs + (i * 256 + tid) * 4) = p;
    }
    __syncthreads();

    // ---- GEMM1: logits(16l x 16n) for this wave's n-subtile; exp; share ----
    {
      v8f acc = {0.f, 0.f, 0.f, 0.f, 0.f, 0.f, 0.f, 0.f};
      const int nrow = wave * 16 + ln;  // B-frag column = n
#pragma unroll
      for (int kc = 0; kc < 8; ++kc) {
        Frag af, bfX;
        // A layout (16x32): lane(hf, row=ln), v0-3: k=8hf+0..7, v4-7: k=16+8hf+0..7
        af.q[0]  = *(const u32x4*)(wa + ln * Dd + kc * 32 + hf * 8);
        af.q[1]  = *(const u32x4*)(wa + ln * Dd + kc * 32 + 16 + hf * 8);
        // X^T as B: v0-3: k=8hf+0..7, v4-7: k=16+8hf+0..7
        bfX.q[0] = *(const u32x4*)(xs + nrow * Dd + kc * 32 + hf * 8);
        bfX.q[1] = *(const u32x4*)(xs + nrow * Dd + kc * 32 + 16 + hf * 8);
        acc = __builtin_amdgcn_wmma_f32_16x16x32_bf16(
            false, af.v, false, bfX.v, (short)0, acc, false, false);
      }
      // D layout: VGPR r -> l = r + 8*hf, col n = ln
#pragma unroll
      for (int r = 0; r < 8; ++r) {
        float e = __expf(acc[r]);
        zacc[r] += e;
        et[(r + 8 * hf) * BLKN + wave * 16 + ln] = f2bf(e);
      }
    }
    __syncthreads();  // E tile complete (written by all waves)

    // ---- GEMM2 (VV tile) + GEMM3 (T += E·VV) for this wave's c-tile ----
    unsigned short* myvv = vvt + wave * (16 * 32);
    for (int p = 0; p < 4; ++p) {        // 32 n per GEMM3
#pragma unroll
      for (int s2 = 0; s2 < 2; ++s2) {
        const int n0 = p * 32 + s2 * 16;
        v8f accv = {0.f, 0.f, 0.f, 0.f, 0.f, 0.f, 0.f, 0.f};
#pragma unroll
        for (int kc = 0; kc < 8; ++kc) {
          Frag xa;  // X as A: lane(hf, m=ln) -> row n0+ln
          xa.q[0] = *(const u32x4*)(xs + (n0 + ln) * Dd + kc * 32 + hf * 16);
          xa.q[1] = *(const u32x4*)(xs + (n0 + ln) * Dd + kc * 32 + hf * 16 + 8);
          accv = __builtin_amdgcn_wmma_f32_16x16x32_bf16(
              false, xa.v, false, wvf[kc].v, (short)0, accv, false, false);
        }
        // D: VGPR r -> n_local = r + 8*hf, col c = ln. Store VV^T[c][n] bf16,
        // 8 consecutive n -> one 16B store.
        unsigned short t[8];
#pragma unroll
        for (int r = 0; r < 8; ++r) t[r] = f2bf(accv[r]);
        u32x4 pk;
        pk.x = (unsigned)t[0] | ((unsigned)t[1] << 16);
        pk.y = (unsigned)t[2] | ((unsigned)t[3] << 16);
        pk.z = (unsigned)t[4] | ((unsigned)t[5] << 16);
        pk.w = (unsigned)t[6] | ((unsigned)t[7] << 16);
        *(u32x4*)(myvv + ln * 32 + s2 * 16 + hf * 8) = pk;
      }
      // GEMM3: A = exp-weights (16l x 32n), B = VV^T scratch (32n x 16c)
      Frag ef, vf;
      ef.q[0] = *(const u32x4*)(et + ln * BLKN + p * 32 + hf * 8);
      ef.q[1] = *(const u32x4*)(et + ln * BLKN + p * 32 + 16 + hf * 8);
      vf.q[0] = *(const u32x4*)(myvv + ln * 32 + hf * 8);
      vf.q[1] = *(const u32x4*)(myvv + ln * 32 + 16 + hf * 8);
      tacc = __builtin_amdgcn_wmma_f32_16x16x32_bf16(
          false, ef.v, false, vf.v, (short)0, tacc, false, false);
    }
    __syncthreads();  // xs / et reused next iteration
  }

  // ---- reduce Z across lanes/waves via LDS, write partials ----
#pragma unroll
  for (int r = 0; r < 8; ++r) atomicAdd(&zl[r + 8 * hf], zacc[r]);
  __syncthreads();

  float* Tout = Tp + (size_t)(b * SPLITS + sp) * Ld * Cd;
#pragma unroll
  for (int r = 0; r < 8; ++r)
    Tout[(r + 8 * hf) * Cd + wave * 16 + ln] = tacc[r];
  if (tid < Ld) Zp[(b * SPLITS + sp) * Ld + tid] = zl[tid];
}

// ---- combine split partials: out = (sum_s num) / (sum_s Z) ----
__global__ __launch_bounds__(256)
void tok_reduce(const float* __restrict__ Tp, const float* __restrict__ Zp,
                float* __restrict__ out) {
  int idx = blockIdx.x * 256 + threadIdx.x;   // [B][L][C] flat
  int c = idx & (Cd - 1);
  int l = (idx >> 7) & (Ld - 1);
  int b = idx >> 11;
  float num = 0.f, z = 0.f;
#pragma unroll
  for (int s = 0; s < SPLITS; ++s) {
    num += Tp[((size_t)(b * SPLITS + s) * Ld + l) * Cd + c];
    z   += Zp[(b * SPLITS + s) * Ld + l];
  }
  out[idx] = num / z;
}

extern "C" void kernel_launch(void* const* d_in, const int* in_sizes, int n_in,
                              void* d_out, int out_size, void* d_ws, size_t ws_size,
                              hipStream_t stream) {
  const float* X  = (const float*)d_in[0];
  const float* wA = (const float*)d_in[1];
  const float* wV = (const float*)d_in[2];

  float* Tp = (float*)d_ws;                               // [B][S][L][C] f32
  float* Zp = Tp + (size_t)Bd * SPLITS * Ld * Cd;         // [B][S][L]    f32

  const size_t lds_bytes =
      (size_t)(128 * 256 + 256 * 128 + 16 * 256 + 16 * 128 + 8 * 16 * 32) * 2 +
      Ld * sizeof(float);  // ~148 KB, fits gfx1250's 320 KB WGP LDS

  tok_main<<<Bd * SPLITS, 256, lds_bytes, stream>>>(X, wA, wV, Tp, Zp);
  tok_reduce<<<(Bd * Ld * Cd) / 256, 256, 0, stream>>>(Tp, Zp, (float*)d_out);
}
